// CausalSelfAttention_37065567764860
// MI455X (gfx1250) — compile-verified, matrix operand order checked by reading, not run on hardware
//
#include <hip/hip_runtime.h>
#include <hip/hip_bf16.h>
#include <stdint.h>

typedef __bf16 bf16;
typedef __bf16 v16bf __attribute__((ext_vector_type(16)));
typedef float  v8f   __attribute__((ext_vector_type(8)));
typedef int    v4i   __attribute__((ext_vector_type(4)));

#define B_   4
#define T_   2048
#define C_   1024
#define H_   16

// ---- probe for the gfx1250 async global->LDS builtin (compile-safe guard) ----
#if defined(__has_builtin)
#  if __has_builtin(__builtin_amdgcn_global_load_async_to_lds_b128)
#    define ASYNC_LDS 1
#  endif
#endif
#ifndef ASYNC_LDS
#  define ASYNC_LDS 0
#endif

#if ASYNC_LDS
#warning "CDNA5: async global_load_async_to_lds_b128 path ENABLED"
// Parameter types per hipcc diagnostic: (int4 AS1*, int4 AS3*, imm offset, imm cpol)
typedef __attribute__((address_space(1))) v4i* g_v4i_p;
typedef __attribute__((address_space(3))) v4i* l_v4i_p;
__device__ __forceinline__ void async_cp16(const bf16* g, bf16* l) {
  __builtin_amdgcn_global_load_async_to_lds_b128((g_v4i_p)g, (l_v4i_p)l, 0, 0);
}
__device__ __forceinline__ void async_wait0() {
#  if __has_builtin(__builtin_amdgcn_s_wait_asynccnt)
  __builtin_amdgcn_s_wait_asynccnt(0);
#  else
  asm volatile("s_wait_asynccnt 0x0" ::: "memory");
#  endif
}
#else
#warning "CDNA5: async LDS path DISABLED (builtin missing; using ds_store staging)"
#endif

// Load 16 contiguous bf16 (32 bytes) as a WMMA fragment (global or LDS).
__device__ __forceinline__ v16bf ld16bf(const bf16* p) {
  union { uint4 u[2]; v16bf v; } t;
  const uint4* q = reinterpret_cast<const uint4*>(p);
  t.u[0] = q[0];
  t.u[1] = q[1];
  return t.v;
}

__device__ __forceinline__ v8f wmma_bf16(v16bf a, v16bf b, v8f c) {
  return __builtin_amdgcn_wmma_f32_16x16x32_bf16(
      false, a, false, b, (short)0, c, false, false);
}

// ---------------- fp32 -> bf16 elementwise (4 per thread) ----------------
__global__ __launch_bounds__(256) void cvt_f32_bf16(const float* __restrict__ src,
                                                    bf16* __restrict__ dst, int n4) {
  int i = blockIdx.x * 256 + threadIdx.x;
  if (i >= n4) return;
  float4 f = reinterpret_cast<const float4*>(src)[i];
  union { bf16 h[4]; uint2 u; } t;
  t.h[0] = (bf16)f.x; t.h[1] = (bf16)f.y; t.h[2] = (bf16)f.z; t.h[3] = (bf16)f.w;
  reinterpret_cast<uint2*>(dst)[i] = t.u;
}

// ---------------- fp32 [K,N] -> bf16 [N,K] (transpose + convert) ----------------
__global__ __launch_bounds__(256) void cvt_transpose_bf16(const float* __restrict__ src,
                                                          bf16* __restrict__ dst,
                                                          int K, int N) {
  int tid = blockIdx.x * 256 + threadIdx.x;
  if (tid >= K * N) return;
  int k = tid % K;
  int n = tid / K;
  dst[tid] = (bf16)src[(size_t)k * N + n];
}

// Chunk mapping for one 128x32 A tile + 128x32 B tile: 1024 chunks of 16B.
// c < 512 -> A, else B; row = (c&511)>>2, 8-elem segment = (c&3)*8.
__device__ __forceinline__ const bf16* tile_src(const bf16* A, const bf16* Bt,
                                                int m_blk, int n_blk, int K,
                                                int k0, int c) {
  const int isB = c >> 9;
  const int rr  = (c & 511) >> 2;
  const int s8  = (c & 3) * 8;
  const bf16* base = isB ? (Bt + (size_t)(n_blk + rr) * K)
                         : (A  + (size_t)(m_blk + rr) * K);
  return base + k0 + s8;
}
__device__ __forceinline__ bf16* tile_dst(bf16 (*As)[40], bf16 (*Bs)[40], int c) {
  const int isB = c >> 9;
  const int rr  = (c & 511) >> 2;
  const int s8  = (c & 3) * 8;
  return isB ? &Bs[rr][s8] : &As[rr][s8];
}

// ---------------- WMMA GEMM: out[M,N] = A[M,K] * Bt[N,K]^T + bias ----------------
// 128 threads = 4 waves; block tile 128x128, BK=32; wave tile 64x64 = 4x4 WMMA.
// Double-buffered LDS, one barrier per K-step, async staging when available.
__global__ __launch_bounds__(128) void gemm_bf16_wmma(
    const bf16* __restrict__ A,     // [M,K] bf16 row-major
    const bf16* __restrict__ Bt,    // [N,K] bf16 row-major (pre-transposed B)
    const float* __restrict__ bias, // [N]
    float* __restrict__ outF, bf16* __restrict__ outH,
    int M, int N, int K, int write_f32)
{
  __shared__ bf16 As[2][128][40];   // +8 pad: 80B rows, 16B aligned
  __shared__ bf16 Bs[2][128][40];

  const int tid  = threadIdx.x;
  const int lane = tid & 31;
  const int wave = tid >> 5;        // 0..3
  const int wm   = wave & 1;        // 2 waves tile M
  const int wn   = wave >> 1;       // 2 waves tile N
  const int m_blk = blockIdx.y * 128;
  const int n_blk = blockIdx.x * 128;

  const int colbase = lane & 15;
  const int kseg    = (lane >> 4) * 16;
  const int half8   = (lane >> 4) * 8;

  v8f acc[4][4];
  #pragma unroll
  for (int mi = 0; mi < 4; ++mi)
    #pragma unroll
    for (int ni = 0; ni < 4; ++ni)
      #pragma unroll
      for (int e = 0; e < 8; ++e) acc[mi][ni][e] = 0.0f;

  const int nIter = K >> 5;

#if ASYNC_LDS
  #pragma unroll
  for (int i = 0; i < 8; ++i) {
    const int c = tid + 128 * i;
    async_cp16(tile_src(A, Bt, m_blk, n_blk, K, 0, c), tile_dst(As[0], Bs[0], c));
  }
  for (int it = 0; it < nIter; ++it) {
    const int p = it & 1;
    async_wait0();
    __syncthreads();
    if (it + 1 < nIter) {
      #pragma unroll
      for (int i = 0; i < 8; ++i) {
        const int c = tid + 128 * i;
        async_cp16(tile_src(A, Bt, m_blk, n_blk, K, (it + 1) << 5, c),
                   tile_dst(As[p ^ 1], Bs[p ^ 1], c));
      }
    }
    v16bf af[4], bfr[4];
    #pragma unroll
    for (int mi = 0; mi < 4; ++mi)
      af[mi] = ld16bf(&As[p][wm * 64 + mi * 16 + colbase][kseg]);
    #pragma unroll
    for (int ni = 0; ni < 4; ++ni)
      bfr[ni] = ld16bf(&Bs[p][wn * 64 + ni * 16 + colbase][kseg]);
    #pragma unroll
    for (int mi = 0; mi < 4; ++mi)
      #pragma unroll
      for (int ni = 0; ni < 4; ++ni)
        acc[mi][ni] = wmma_bf16(af[mi], bfr[ni], acc[mi][ni]);
  }
#else
  uint4 rg[8];
  #pragma unroll
  for (int i = 0; i < 8; ++i)
    rg[i] = *reinterpret_cast<const uint4*>(tile_src(A, Bt, m_blk, n_blk, K, 0, tid + 128 * i));
  for (int it = 0; it < nIter; ++it) {
    const int p = it & 1;
    #pragma unroll
    for (int i = 0; i < 8; ++i)
      *reinterpret_cast<uint4*>(tile_dst(As[p], Bs[p], tid + 128 * i)) = rg[i];
    __syncthreads();
    if (it + 1 < nIter) {
      #pragma unroll
      for (int i = 0; i < 8; ++i)
        rg[i] = *reinterpret_cast<const uint4*>(
            tile_src(A, Bt, m_blk, n_blk, K, (it + 1) << 5, tid + 128 * i));
    }
    v16bf af[4], bfr[4];
    #pragma unroll
    for (int mi = 0; mi < 4; ++mi)
      af[mi] = ld16bf(&As[p][wm * 64 + mi * 16 + colbase][kseg]);
    #pragma unroll
    for (int ni = 0; ni < 4; ++ni)
      bfr[ni] = ld16bf(&Bs[p][wn * 64 + ni * 16 + colbase][kseg]);
    #pragma unroll
    for (int mi = 0; mi < 4; ++mi)
      #pragma unroll
      for (int ni = 0; ni < 4; ++ni)
        acc[mi][ni] = wmma_bf16(af[mi], bfr[ni], acc[mi][ni]);
  }
#endif

  #pragma unroll
  for (int mi = 0; mi < 4; ++mi) {
    #pragma unroll
    for (int ni = 0; ni < 4; ++ni) {
      const int col = n_blk + wn * 64 + ni * 16 + colbase;
      const float bv = bias[col];
      #pragma unroll
      for (int i = 0; i < 8; ++i) {
        const int row = m_blk + wm * 64 + mi * 16 + i + half8;
        const float v = acc[mi][ni][i] + bv;
        const size_t idx = (size_t)row * N + col;
        if (write_f32) outF[idx] = v;
        else           outH[idx] = (bf16)v;
      }
    }
  }
}

// ---------------- Flash attention (causal), bf16 WMMA, fp32 softmax ----------------
// grid: (T/128, B*H). 8 waves; wave owns 16 query rows. Key tiles of 64.
__global__ __launch_bounds__(256) void attn_fwd_wmma(
    const bf16* __restrict__ qkv,   // [B*T, 3072] : [q|k|v] each 1024
    bf16* __restrict__ out)         // [B*T, 1024]
{
  __shared__ bf16 Ks[64][72];        // K tile, row-major (keys x d)
  __shared__ bf16 Vt[64][72];        // V tile, TRANSPOSED (d x keys)
  __shared__ bf16 Ps[8][16][72];     // per-wave P staging (C-layout -> A-fragment)

  const int tid  = threadIdx.x;
  const int lane = tid & 31;
  const int wave = tid >> 5;
  const int qt   = blockIdx.x;
  const int bh   = blockIdx.y;
  const int b    = bh >> 4;
  const int h    = bh & 15;

  const int colbase = lane & 15;
  const int kseg    = (lane >> 4) * 16;
  const int half8   = (lane >> 4) * 8;

  const int qw = qt * 128 + wave * 16;

  const size_t rowQ = (size_t)(b * T_ + qw + colbase) * 3072 + h * 64;
  v16bf aq[2];
  #pragma unroll
  for (int s = 0; s < 2; ++s) aq[s] = ld16bf(qkv + rowQ + s * 32 + kseg);

  v8f o_acc[4];
  #pragma unroll
  for (int j = 0; j < 4; ++j)
    #pragma unroll
    for (int e = 0; e < 8; ++e) o_acc[j][e] = 0.0f;
  float m_run[8], l_run[8];
  #pragma unroll
  for (int i = 0; i < 8; ++i) { m_run[i] = -1e30f; l_run[i] = 0.0f; }

  const int nkt = 2 * qt + 2;
  const int r   = tid >> 2;        // 0..63 row for V staging
  const int seg = (tid & 3) * 16;

  for (int kt = 0; kt < nkt; ++kt) {
    const size_t rowKV = (size_t)(b * T_ + kt * 64 + r) * 3072 + h * 64;
#if ASYNC_LDS
    // K tile via async direct-to-LDS: 2 chunks of 16B per thread.
    #pragma unroll
    for (int i = 0; i < 2; ++i) {
      const int c  = tid + 256 * i;
      const int rr = c >> 3;
      const int s8 = (c & 7) * 8;
      async_cp16(qkv + (size_t)(b * T_ + kt * 64 + rr) * 3072 + 1024 + h * 64 + s8,
                 &Ks[rr][s8]);
    }
#else
    {
      const uint4* gk = reinterpret_cast<const uint4*>(qkv + rowKV + 1024 + seg);
      uint4 k0 = gk[0], k1 = gk[1];
      *reinterpret_cast<uint4*>(&Ks[r][seg])     = k0;
      *reinterpret_cast<uint4*>(&Ks[r][seg + 8]) = k1;
    }
#endif
    // V tile: transpose during staging (element scatter, stays on ds path).
    {
      union { uint4 u[2]; bf16 e[16]; } vv;
      const uint4* gv = reinterpret_cast<const uint4*>(qkv + rowKV + 2048 + seg);
      vv.u[0] = gv[0]; vv.u[1] = gv[1];
      #pragma unroll
      for (int j = 0; j < 16; ++j) Vt[seg + j][r] = vv.e[j];
    }
#if ASYNC_LDS
    async_wait0();
#endif
    __syncthreads();

    // ---- S = Q * K^T (16 x 64) ----
    v8f s_acc[4];
    #pragma unroll
    for (int j = 0; j < 4; ++j)
      #pragma unroll
      for (int e = 0; e < 8; ++e) s_acc[j][e] = 0.0f;
    #pragma unroll
    for (int s = 0; s < 2; ++s)
      #pragma unroll
      for (int j = 0; j < 4; ++j)
        s_acc[j] = wmma_bf16(aq[s],
                             ld16bf(&Ks[j * 16 + colbase][s * 32 + kseg]),
                             s_acc[j]);

    // ---- causal mask + online softmax ----
    #pragma unroll
    for (int i = 0; i < 8; ++i) {
      const int qg = qw + i + half8;
      float mx = -1e30f;
      #pragma unroll
      for (int j = 0; j < 4; ++j) {
        const int kg = kt * 64 + j * 16 + colbase;
        float v = s_acc[j][i] * 0.125f;
        v = (kg <= qg) ? v : -1e30f;
        s_acc[j][i] = v;
        mx = fmaxf(mx, v);
      }
      #pragma unroll
      for (int off = 1; off < 16; off <<= 1)
        mx = fmaxf(mx, __shfl_xor(mx, off, 16));
      const float mn = fmaxf(m_run[i], mx);
      const float al = __expf(m_run[i] - mn);
      m_run[i] = mn;
      float rs = 0.0f;
      #pragma unroll
      for (int j = 0; j < 4; ++j) {
        const float p = __expf(s_acc[j][i] - mn);
        s_acc[j][i] = p;
        rs += p;
      }
      #pragma unroll
      for (int off = 1; off < 16; off <<= 1)
        rs += __shfl_xor(rs, off, 16);
      l_run[i] = l_run[i] * al + rs;
      #pragma unroll
      for (int j = 0; j < 4; ++j) o_acc[j][i] *= al;
    }

    // ---- P: C-layout -> A-fragment layout via per-wave LDS buffer ----
    #pragma unroll
    for (int i = 0; i < 8; ++i)
      #pragma unroll
      for (int j = 0; j < 4; ++j)
        Ps[wave][i + half8][j * 16 + colbase] = (bf16)s_acc[j][i];
    v16bf ap[2];
    #pragma unroll
    for (int s = 0; s < 2; ++s)
      ap[s] = ld16bf(&Ps[wave][colbase][s * 32 + kseg]);

    // ---- O += P * V ----
    #pragma unroll
    for (int s = 0; s < 2; ++s)
      #pragma unroll
      for (int j = 0; j < 4; ++j)
        o_acc[j] = wmma_bf16(ap[s],
                             ld16bf(&Vt[j * 16 + colbase][s * 32 + kseg]),
                             o_acc[j]);

    __syncthreads();
  }

  #pragma unroll
  for (int i = 0; i < 8; ++i) {
    const float inv = 1.0f / l_run[i];
    const int tq = qw + i + half8;
    #pragma unroll
    for (int j = 0; j < 4; ++j) {
      const int d = j * 16 + colbase;
      out[(size_t)(b * T_ + tq) * 1024 + h * 64 + d] = (bf16)(o_acc[j][i] * inv);
    }
  }
}

extern "C" void kernel_launch(void* const* d_in, const int* in_sizes, int n_in,
                              void* d_out, int out_size, void* d_ws, size_t ws_size,
                              hipStream_t stream) {
  const float* x     = (const float*)d_in[0];
  const float* w_qkv = (const float*)d_in[1];
  const float* b_qkv = (const float*)d_in[2];
  const float* w_out = (const float*)d_in[3];
  const float* b_out = (const float*)d_in[4];
  float* out = (float*)d_out;

  char* ws = (char*)d_ws;
  const size_t BT = (size_t)B_ * T_;                        // 8192 tokens
  bf16* xh    = (bf16*)(ws);                                // [8192,1024]
  bf16* wqkvT = (bf16*)(ws + (BT * C_) * 2);                // [3072,1024]
  bf16* woutT = (bf16*)(ws + (BT * C_ + 3ull*C_*C_) * 2);   // [1024,1024]
  bf16* qkvh  = (bf16*)(ws + (BT * C_ + 4ull*C_*C_) * 2);   // [8192,3072]
  bf16* atth  = (bf16*)(ws + (BT * C_ + 4ull*C_*C_ + BT*3ull*C_) * 2); // [8192,1024]

  cvt_f32_bf16<<<(int)((BT * C_ / 4 + 255) / 256), 256, 0, stream>>>(x, xh, (int)(BT * C_ / 4));
  cvt_transpose_bf16<<<(3 * C_ * C_ + 255) / 256, 256, 0, stream>>>(w_qkv, wqkvT, C_, 3 * C_);
  cvt_transpose_bf16<<<(C_ * C_ + 255) / 256, 256, 0, stream>>>(w_out, woutT, C_, C_);

  gemm_bf16_wmma<<<dim3(3 * C_ / 128, (int)(BT / 128)), 128, 0, stream>>>(
      xh, wqkvT, b_qkv, nullptr, qkvh, (int)BT, 3 * C_, C_, 0);

  attn_fwd_wmma<<<dim3(T_ / 128, B_ * H_), 256, 0, stream>>>(qkvh, atth);

  gemm_bf16_wmma<<<dim3(C_ / 128, (int)(BT / 128)), 128, 0, stream>>>(
      atth, woutT, b_out, out, nullptr, (int)BT, C_, C_, 1);
}
